// SHViTBlockWithDynamicSelect_75058848465302
// MI455X (gfx1250) — matrix operand order repaired and checked by reading, not compiled, not run on hardware
//
#include <hip/hip_runtime.h>
#include <hip/hip_bf16.h>
#include <math.h>

// Problem constants
#define B_   16
#define C_   256
#define HW_  1024      // 32*32
#define AD_  16
#define SC_  32
#define HID_ 1024      // 4*C

// FFN column tile (spatial positions per block)
#define NCOLS 32

typedef _Float16 half_t;
typedef __attribute__((ext_vector_type(16))) _Float16 v16h;
typedef __attribute__((ext_vector_type(8)))  float    v8f;

// CDNA5 WMMA f16 A-fragment K index for half h (0..15), lane-high bit hi.
// Contiguous order: K = h + 16*hi (matches B-fragment fills)
__device__ __forceinline__ int afrag_k(int h, int hi) {
  return h + 16 * hi;
}

// ---------------------------------------------------------------------------
// 1) Squeeze-Excite: channel means -> fc1+relu -> fc2+sigmoid -> scale[B_,C_]
// ---------------------------------------------------------------------------
__global__ __launch_bounds__(256)
void k_se(const float* __restrict__ x,
          const float* __restrict__ fc1_w, const float* __restrict__ fc1_b,
          const float* __restrict__ fc2_w, const float* __restrict__ fc2_b,
          float* __restrict__ scale) {
  int b = blockIdx.x;
  int c = threadIdx.x;               // 256 threads = 256 channels
  __shared__ float mean[C_];
  __shared__ float s1[C_ / 4];       // 64
  const float* xb = x + (size_t)b * C_ * HW_;
  float acc = 0.f;
  const float* row = xb + (size_t)c * HW_;
  for (int n = 0; n < HW_; n += 4) {
    acc += row[n] + row[n + 1] + row[n + 2] + row[n + 3];
  }
  mean[c] = acc * (1.0f / HW_);
  __syncthreads();
  if (c < 64) {
    float a = fc1_b[c];
    for (int k = 0; k < C_; ++k) a += fc1_w[c * C_ + k] * mean[k];
    s1[c] = fmaxf(a, 0.f);
  }
  __syncthreads();
  float a = fc2_b[c];
  for (int k = 0; k < 64; ++k) a += fc2_w[c * 64 + k] * s1[k];
  scale[b * C_ + c] = 1.f / (1.f + __expf(-a));
}

// ---------------------------------------------------------------------------
// 2) f32 -> f16 weight conversion (one-time per launch, tiny)
// ---------------------------------------------------------------------------
__global__ __launch_bounds__(256)
void k_cvt(const float* __restrict__ src, half_t* __restrict__ dst, int n) {
  int i = blockIdx.x * blockDim.x + threadIdx.x;
  if (i < n) dst[i] = (half_t)src[i];
}

// ---------------------------------------------------------------------------
// 3) QKV: per batch (64 x 32) @ (32 x 1024), SE scale fused into B-fragment.
//    One v_wmma per 16x16 tile (K = 32 exactly). Output f16: qkvh[b][64][1024]
// ---------------------------------------------------------------------------
__global__ __launch_bounds__(256)
void k_qkv(const float* __restrict__ x, const float* __restrict__ scale,
           const half_t* __restrict__ wq, const float* __restrict__ qkv_b,
           half_t* __restrict__ qkvh) {
  int b    = blockIdx.x;
  int nblk = blockIdx.y;                 // 8 blocks of 128 columns
  int wave = threadIdx.x >> 5;
  int lane = threadIdx.x & 31;
  int l15 = lane & 15, hi = lane >> 4;
  int nbase = nblk * 128 + wave * 16;
  const float* xb = x + (size_t)b * C_ * HW_;
  const float* sb = scale + b * C_;
  half_t* outb = qkvh + (size_t)b * 64 * HW_;

  // B-fragment: xs[k][n] = x[b][k][n] * scale[b][k], k = 0..31
  v16h bf;
  int n = nbase + l15;
#pragma unroll
  for (int h = 0; h < 16; ++h) {
    int k = h + 16 * hi;
    bf[h] = (half_t)(xb[(size_t)k * HW_ + n] * sb[k]);
  }

#pragma unroll
  for (int mt = 0; mt < 4; ++mt) {
    v16h af;
    int m = mt * 16 + l15;
#pragma unroll
    for (int h = 0; h < 16; ++h) af[h] = wq[m * SC_ + afrag_k(h, hi)];
    v8f d = {};
    d = __builtin_amdgcn_wmma_f32_16x16x32_f16(false, af, false, bf,
                                               (short)0, d, false, false);
#pragma unroll
    for (int r = 0; r < 8; ++r) {
      int mo = mt * 16 + r + 8 * hi;
      outb[(size_t)mo * HW_ + n] = (half_t)(d[r] + qkv_b[mo]);
    }
  }
}

// ---------------------------------------------------------------------------
// 4) Attention + proj, per (batch, 16-query tile). 8 waves/block.
//    Phase 1: S = (Q^T K) * 0.25 via WMMA (K=16 zero-padded to 32) -> LDS f16
//    Phase 2: softmax over 1024 (LDS)
//    Phase 3: O = V @ P^T via WMMA, K split across waves, LDS f32 reduce
//    Phase 4: 32x32 proj (+bias), write y channels [0,32)
// ---------------------------------------------------------------------------
__global__ __launch_bounds__(256)
void k_attn(const half_t* __restrict__ qkvh,
            const float* __restrict__ proj_w, const float* __restrict__ proj_b,
            float* __restrict__ y) {
  int b  = blockIdx.x;
  int nt = blockIdx.y;                   // 64 query tiles of 16
  int tid = threadIdx.x;
  int wave = tid >> 5, lane = tid & 31, l15 = lane & 15, hi = lane >> 4;

  __shared__ half_t qA[16 * 16];         // q^T tile  [n_local][c]
  __shared__ half_t Ph[16 * 1024];       // scores / probs [n_local][m]  (32KB)
  __shared__ float  red[16 * 16];
  __shared__ float  rowv[16];
  __shared__ float  Oacc[32 * 16];       // [c][n_local]

  const half_t* qk = qkvh + (size_t)b * 64 * HW_;

  // load q^T tile (channels 0..15, columns nt*16..+15)
  {
    int nl = tid >> 4, c = tid & 15;     // exactly 256 threads
    qA[nl * 16 + c] = qk[(size_t)c * HW_ + nt * 16 + nl];
  }
  __syncthreads();

  // A-fragment (shared across all m-tiles of this wave), K 16..31 zero-padded
  v16h af;
#pragma unroll
  for (int h = 0; h < 16; ++h) {
    int k = afrag_k(h, hi);
    af[h] = (k < AD_) ? qA[l15 * 16 + k] : (half_t)0.f;
  }

  // Phase 1: scores for columns [wave*128, wave*128+128)
  for (int mt = 0; mt < 8; ++mt) {
    int mbase = wave * 128 + mt * 16;
    v16h bf;
#pragma unroll
    for (int h = 0; h < 16; ++h) {
      int k = h + 16 * hi;
      bf[h] = (k < AD_) ? qk[(size_t)(AD_ + k) * HW_ + mbase + l15]
                        : (half_t)0.f;
    }
    v8f d = {};
    d = __builtin_amdgcn_wmma_f32_16x16x32_f16(false, af, false, bf,
                                               (short)0, d, false, false);
#pragma unroll
    for (int r = 0; r < 8; ++r) {
      int row = r + 8 * hi;
      Ph[row * 1024 + mbase + l15] = (half_t)(d[r] * 0.25f); // AD^-0.5
    }
  }
  __syncthreads();

  // Phase 2: softmax over 1024 per row; 16 threads per row
  {
    int row = tid >> 4, sub = tid & 15;
    float mx = -1e30f;
    for (int j = sub; j < 1024; j += 16)
      mx = fmaxf(mx, (float)Ph[row * 1024 + j]);
    red[row * 16 + sub] = mx;
    __syncthreads();
    if (sub == 0) {
      float m2 = red[row * 16];
      for (int t = 1; t < 16; ++t) m2 = fmaxf(m2, red[row * 16 + t]);
      rowv[row] = m2;
    }
    __syncthreads();
    float rm = rowv[row];
    float sum = 0.f;
    for (int j = sub; j < 1024; j += 16) {
      float e = __expf((float)Ph[row * 1024 + j] - rm);
      Ph[row * 1024 + j] = (half_t)e;
      sum += e;
    }
    red[row * 16 + sub] = sum;
    __syncthreads();
    if (sub == 0) {
      float s2 = 0.f;
      for (int t = 0; t < 16; ++t) s2 += red[row * 16 + t];
      rowv[row] = 1.f / s2;
    }
    __syncthreads();
    float inv = rowv[row];
    for (int j = sub; j < 1024; j += 16)
      Ph[row * 1024 + j] = (half_t)((float)Ph[row * 1024 + j] * inv);
  }

  // zero Oacc (512 floats)
  Oacc[tid] = 0.f;
  if (tid < 256) Oacc[tid + 256] = 0.f;
  __syncthreads();

  // Phase 3: O[c][n] = sum_m V[c][m] * P[n][m]; wave handles K-range of 128
#pragma unroll
  for (int ct = 0; ct < 2; ++ct) {
    v8f d = {};
    for (int ks = 0; ks < 4; ++ks) {
      int kb = wave * 128 + ks * 32;
      v16h av, bp;
#pragma unroll
      for (int h = 0; h < 16; ++h) {
        int k = kb + afrag_k(h, hi);
        av[h] = qk[(size_t)(2 * AD_ + ct * 16 + l15) * HW_ + k];
      }
#pragma unroll
      for (int h = 0; h < 16; ++h) {
        int k = kb + h + 16 * hi;
        bp[h] = Ph[l15 * 1024 + k];
      }
      d = __builtin_amdgcn_wmma_f32_16x16x32_f16(false, av, false, bp,
                                                 (short)0, d, false, false);
    }
#pragma unroll
    for (int r = 0; r < 8; ++r) {
      int c = ct * 16 + r + 8 * hi;
      atomicAdd(&Oacc[c * 16 + l15], d[r]);   // ds_add_f32 cross-wave reduce
    }
  }
  __syncthreads();

  // Phase 4: proj (32x32) + bias, write y channels [0,32)
  for (int t = tid; t < 512; t += 256) {
    int o = t >> 4, n = t & 15;
    float acc = proj_b[o];
    for (int c = 0; c < 32; ++c) acc += proj_w[o * 32 + c] * Oacc[c * 16 + n];
    y[((size_t)b * C_ + o) * HW_ + nt * 16 + n] = acc;
  }
}

// ---------------------------------------------------------------------------
// 5) y channels [32,256) = x * scale (elementwise)
// ---------------------------------------------------------------------------
__global__ __launch_bounds__(256)
void k_scale_rest(const float* __restrict__ x, const float* __restrict__ scale,
                  float* __restrict__ y) {
  size_t i = (size_t)blockIdx.x * blockDim.x + threadIdx.x;
  const size_t total = (size_t)B_ * (C_ - SC_) * HW_;
  if (i >= total) return;
  int n = (int)(i % HW_);
  size_t t = i / HW_;
  int c = (int)(t % (C_ - SC_)) + SC_;
  int b = (int)(t / (C_ - SC_));
  size_t off = ((size_t)b * C_ + c) * HW_ + n;
  y[off] = x[off] * scale[b * C_ + c];
}

// ---------------------------------------------------------------------------
// 6) Fused FFN per (batch, 32-column tile), dynamic LDS (80KB of the WGP's
//    320KB; 4 blocks/WGP). Two N-tile accumulators per A-fragment load halve
//    the W1/W2 L2 re-stream (~512MB total vs 1GB with 16-col tiles).
//    h = relu(W1 @ y_tile + b1) kept as f16 in LDS, then out = W2 @ h + b2.
// ---------------------------------------------------------------------------
__global__ __launch_bounds__(256)
void k_ffn(const float* __restrict__ y, const half_t* __restrict__ w1h,
           const float* __restrict__ b1, const half_t* __restrict__ w2h,
           const float* __restrict__ b2, float* __restrict__ out) {
  extern __shared__ char smem[];
  half_t* ys = (half_t*)smem;                       // [NCOLS][256]  16KB
  half_t* hs = (half_t*)(smem + NCOLS * C_ * 2);    // [NCOLS][1024] 64KB

  int b    = blockIdx.x;
  int nblk = blockIdx.y;                  // 32 tiles of NCOLS columns
  int tid = threadIdx.x, wave = tid >> 5, lane = tid & 31;
  int l15 = lane & 15, hi = lane >> 4;
  int ncol = nblk * NCOLS;

  // stage y tile -> f16 LDS (transposed: [n][k], contiguous-K for B-frags)
  for (int idx = tid; idx < NCOLS * C_; idx += 256) {
    int n = idx & (NCOLS - 1), k = idx >> 5;        // NCOLS == 32
    ys[n * C_ + k] = (half_t)y[((size_t)b * C_ + k) * HW_ + ncol + n];
  }
  __syncthreads();

  // Phase 1: hidden = relu(W1 @ y + b1); each wave owns 128 hidden rows.
  // One A-fragment feeds two N-tiles (acc0/acc1).
  for (int mt = 0; mt < 8; ++mt) {
    int mbase = wave * 128 + mt * 16;
    // prefetch next M-tile's weight rows into cache one iteration ahead
    if (mt < 7)
      __builtin_prefetch(&w1h[(size_t)(mbase + 16 + l15) * C_], 0, 1);
    v8f acc0 = {}, acc1 = {};
    for (int ks = 0; ks < 8; ++ks) {       // K = 256
      int kb = ks * 32;
      v16h af, bf0, bf1;
#pragma unroll
      for (int h = 0; h < 16; ++h)
        af[h] = w1h[(size_t)(mbase + l15) * C_ + kb + afrag_k(h, hi)];
#pragma unroll
      for (int h = 0; h < 16; ++h) {
        int k = kb + h + 16 * hi;
        bf0[h] = ys[l15 * C_ + k];
        bf1[h] = ys[(16 + l15) * C_ + k];
      }
      acc0 = __builtin_amdgcn_wmma_f32_16x16x32_f16(false, af, false, bf0,
                                                    (short)0, acc0, false, false);
      acc1 = __builtin_amdgcn_wmma_f32_16x16x32_f16(false, af, false, bf1,
                                                    (short)0, acc1, false, false);
    }
#pragma unroll
    for (int r = 0; r < 8; ++r) {
      int m = mbase + r + 8 * hi;
      float bias = b1[m];
      hs[l15 * HID_ + m]        = (half_t)fmaxf(acc0[r] + bias, 0.f);
      hs[(16 + l15) * HID_ + m] = (half_t)fmaxf(acc1[r] + bias, 0.f);
    }
  }
  __syncthreads();

  // Phase 2: out = W2 @ hidden + b2; each wave owns 32 output rows.
#pragma unroll
  for (int mt = 0; mt < 2; ++mt) {
    int mbase = wave * 32 + mt * 16;
    if (mt == 0)
      __builtin_prefetch(&w2h[(size_t)(mbase + 16 + l15) * HID_], 0, 1);
    v8f acc0 = {}, acc1 = {};
    for (int ks = 0; ks < 32; ++ks) {      // K = 1024
      int kb = ks * 32;
      v16h af, bf0, bf1;
#pragma unroll
      for (int h = 0; h < 16; ++h)
        af[h] = w2h[(size_t)(mbase + l15) * HID_ + kb + afrag_k(h, hi)];
#pragma unroll
      for (int h = 0; h < 16; ++h) {
        int k = kb + h + 16 * hi;
        bf0[h] = hs[l15 * HID_ + k];
        bf1[h] = hs[(16 + l15) * HID_ + k];
      }
      acc0 = __builtin_amdgcn_wmma_f32_16x16x32_f16(false, af, false, bf0,
                                                    (short)0, acc0, false, false);
      acc1 = __builtin_amdgcn_wmma_f32_16x16x32_f16(false, af, false, bf1,
                                                    (short)0, acc1, false, false);
    }
#pragma unroll
    for (int r = 0; r < 8; ++r) {
      int m = mbase + r + 8 * hi;
      float bias = b2[m];
      out[((size_t)b * C_ + m) * HW_ + ncol + l15]      = acc0[r] + bias;
      out[((size_t)b * C_ + m) * HW_ + ncol + 16 + l15] = acc1[r] + bias;
    }
  }
}

// ---------------------------------------------------------------------------
extern "C" void kernel_launch(void* const* d_in, const int* in_sizes, int n_in,
                              void* d_out, int out_size, void* d_ws, size_t ws_size,
                              hipStream_t stream) {
  (void)in_sizes; (void)n_in; (void)out_size; (void)ws_size;
  const float* x      = (const float*)d_in[0];
  const float* fc1_w  = (const float*)d_in[1];
  const float* fc1_b  = (const float*)d_in[2];
  const float* fc2_w  = (const float*)d_in[3];
  const float* fc2_b  = (const float*)d_in[4];
  const float* qkv_w  = (const float*)d_in[5];
  const float* qkv_b  = (const float*)d_in[6];
  const float* proj_w = (const float*)d_in[7];
  const float* proj_b = (const float*)d_in[8];
  const float* ffn1_w = (const float*)d_in[9];
  const float* ffn1_b = (const float*)d_in[10];
  const float* ffn2_w = (const float*)d_in[11];
  const float* ffn2_b = (const float*)d_in[12];
  float* out = (float*)d_out;

  // Workspace layout (~19 MB total)
  char* ws = (char*)d_ws;
  float*  scale = (float*)ws;   ws += (size_t)B_ * C_ * sizeof(float);
  half_t* qkvh  = (half_t*)ws;  ws += (size_t)B_ * 64 * HW_ * sizeof(half_t);
  float*  yb    = (float*)ws;   ws += (size_t)B_ * C_ * HW_ * sizeof(float);
  half_t* w1h   = (half_t*)ws;  ws += (size_t)HID_ * C_ * sizeof(half_t);
  half_t* w2h   = (half_t*)ws;  ws += (size_t)C_ * HID_ * sizeof(half_t);
  half_t* qwh   = (half_t*)ws;  ws += (size_t)64 * SC_ * sizeof(half_t);

  k_se<<<B_, 256, 0, stream>>>(x, fc1_w, fc1_b, fc2_w, fc2_b, scale);

  k_cvt<<<(HID_ * C_ + 255) / 256, 256, 0, stream>>>(ffn1_w, w1h, HID_ * C_);
  k_cvt<<<(C_ * HID_ + 255) / 256, 256, 0, stream>>>(ffn2_w, w2h, C_ * HID_);
  k_cvt<<<(64 * SC_ + 255) / 256, 256, 0, stream>>>(qkv_w, qwh, 64 * SC_);

  k_qkv<<<dim3(B_, 8), 256, 0, stream>>>(x, scale, qwh, qkv_b, qkvh);

  k_scale_rest<<<(int)(((size_t)B_ * (C_ - SC_) * HW_ + 255) / 256), 256, 0, stream>>>(
      x, scale, yb);

  k_attn<<<dim3(B_, 64), 256, 0, stream>>>(qkvh, proj_w, proj_b, yb);

  // 80KB dynamic LDS: [NCOLS][256] y-tile (f16) + [NCOLS][1024] hidden (f16)
  size_t ffn_lds = (size_t)NCOLS * C_ * 2 + (size_t)NCOLS * HID_ * 2;
  k_ffn<<<dim3(B_, HW_ / NCOLS), 256, ffn_lds, stream>>>(
      yb, w1h, ffn1_b, w2h, ffn2_b, out);
}
